// AttentionWithMoLE_33294586478886
// MI455X (gfx1250) — matrix operand order repaired and checked by reading, not compiled, not run on hardware
//
#include <hip/hip_runtime.h>
#include <hip/hip_bf16.h>
#include <math.h>

typedef __attribute__((ext_vector_type(16))) __bf16 v16bfT;
typedef __attribute__((ext_vector_type(8)))  float  v8fT;
typedef __attribute__((ext_vector_type(4)))  unsigned int u32x4;
typedef __attribute__((ext_vector_type(8)))  unsigned int u32x8;

// ---------------------------------------------------------------------------
// Tensor Data Mover: DMA a rows x cols fp32 tile (row-major, leading dim ld
// in elements) from global memory into LDS at byte offset lds_off.
// D# group0: count=1 | lds_addr | global_addr | type=2.
// D# group1: data_size=4B | tensor_dim0=cols | tensor_dim1=rows |
//            tile_dim0=cols | tile_dim1=rows | tensor_dim0_stride=ld.
// Issued once per wave (EXEC ignored by TDM); tracked by TENSORcnt.
// ---------------------------------------------------------------------------
__device__ __forceinline__ void tdm_load_tile_f32(unsigned lds_off,
                                                  const float* gsrc,
                                                  unsigned rows, unsigned cols,
                                                  unsigned ld)
{
    const unsigned long long ga = (unsigned long long)(size_t)gsrc;
    u32x4 g0;
    g0[0] = 1u;                                        // count=1 (valid), user mode
    g0[1] = lds_off;                                   // LDS byte address
    g0[2] = (unsigned)(ga & 0xffffffffu);              // global_addr[31:0]
    g0[3] = (unsigned)((ga >> 32) & 0x01ffffffu)       // global_addr[56:32]
          | (2u << 30);                                // type=2 ("image")
    u32x8 g1;
    g1[0] = 2u << 16;                                  // data_size=2 -> 4 bytes
    g1[1] = cols << 16;                                // tensor_dim0[15:0]
    g1[2] = rows << 16;                                // tensor_dim0[31:16]=0 | tensor_dim1[15:0]
    g1[3] = cols << 16;                                // tensor_dim1[31:16]=0 | tile_dim0
    g1[4] = rows;                                      // tile_dim1 | tile_dim2=0
    g1[5] = ld;                                        // tensor_dim0_stride[31:0]
    g1[6] = 0u;                                        // stride[47:32]=0 | dim1_stride lo
    g1[7] = 0u;
    asm volatile("tensor_load_to_lds %0, %1" :: "s"(g0), "s"(g1) : "memory");
}

// ---------------------------------------------------------------------------
// Generic bf16-WMMA GEMM: C[M,N] = A[M,K] @ B[K,N], fp32 in/out, bf16 compute.
// Workgroup tile 64x64, 8 waves (wave32), each wave 16x32 (two 16x16 frags).
// Global->LDS staging via the Tensor Data Mover (wave 0 issues, TENSORcnt
// waits, workgroup barrier publishes). Requires M%64==0, K%32==0; N guarded
// via a block-uniform manual fallback for partial N tiles (rank-16 LoRA GEMM).
// ---------------------------------------------------------------------------
__global__ __launch_bounds__(256)
void gemm_bf16_wmma(const float* __restrict__ A, const float* __restrict__ B,
                    float* __restrict__ C, int M, int N, int K,
                    int lda, int ldb, int ldc)
{
    __shared__ float Asf[64 * 32];        // row-major 64x32 (8 KB)
    __shared__ float Bsf[32 * 64];        // row-major 32x64 (8 KB)
    const int tid  = threadIdx.x;
    const int wave = tid >> 5;
    const int lane = tid & 31;
    const int m0 = blockIdx.y * 64;
    const int n0 = blockIdx.x * 64;
    const int wm = (wave >> 1) * 16;      // 0,16,32,48
    const int wn = (wave & 1) * 32;       // 0,32
    const int lm  = lane & 15;
    const int kba = (lane >> 4) * 16;     // A-frag K base (contiguous layout)
    const int kbb = (lane >> 4) * 16;     // B-frag K base (contiguous layout)
    const bool fullN = (n0 + 64 <= N);    // block-uniform

    const unsigned asoff = (unsigned)(size_t)(&Asf[0]);
    const unsigned bsoff = (unsigned)(size_t)(&Bsf[0]);

    v8fT acc0 = {}; v8fT acc1 = {};

    for (int k0 = 0; k0 < K; k0 += 32) {
        // --- stage tiles: TDM from wave 0 ---
        if (wave == 0) {
            tdm_load_tile_f32(asoff, A + (size_t)m0 * lda + k0, 64, 32, (unsigned)lda);
            if (fullN)
                tdm_load_tile_f32(bsoff, B + (size_t)k0 * ldb + n0, 32, 64, (unsigned)ldb);
            __builtin_amdgcn_s_wait_tensorcnt(0);
        }
        if (!fullN) {
            // manual staging with zero fill (uniform per-thread predicate)
            const int r = tid >> 3;            // k: 0..31
            const int c = (tid & 7) * 8;       // n group: 0..56
            float4 v0 = make_float4(0.f, 0.f, 0.f, 0.f), v1 = v0;
            if (n0 + c + 8 <= N) {
                const float* src = B + (size_t)(k0 + r) * ldb + n0 + c;
                v0 = *(const float4*)(src);
                v1 = *(const float4*)(src + 4);
            }
            float* dst = &Bsf[r * 64 + c];
            *(float4*)(dst)     = v0;
            *(float4*)(dst + 4) = v1;
        }
        __syncthreads();

        // --- build fragments (fp32 LDS -> bf16 regs) ---
        v16bfT af, bf0, bf1;
        const float* ap = &Asf[(wm + lm) * 32];
        #pragma unroll
        for (int i = 0; i < 8; ++i) {
            af[i]     = (__bf16)ap[kba + i];
            af[i + 8] = (__bf16)ap[kba + 8 + i];
        }
        #pragma unroll
        for (int i = 0; i < 16; ++i) {
            bf0[i] = (__bf16)Bsf[(kbb + i) * 64 + wn + lm];
            bf1[i] = (__bf16)Bsf[(kbb + i) * 64 + wn + 16 + lm];
        }

        acc0 = __builtin_amdgcn_wmma_f32_16x16x32_bf16(false, af, false, bf0,
                                                       (short)0, acc0, false, false);
        acc1 = __builtin_amdgcn_wmma_f32_16x16x32_bf16(false, af, false, bf1,
                                                       (short)0, acc1, false, false);
        __syncthreads();
    }

    // --- store C frags: VGPR r -> (m = r + 8*(lane>>4), n = lane&15) ---
    const int mbase = m0 + wm + (lane >> 4) * 8;
    const int nc0   = n0 + wn + lm;
    #pragma unroll
    for (int r = 0; r < 8; ++r) {
        const size_t mrow = (size_t)(mbase + r) * ldc;
        if (nc0 < N)      C[mrow + nc0]      = acc0[r];
        if (nc0 + 16 < N) C[mrow + nc0 + 16] = acc1[r];
    }
}

// ---------------------------------------------------------------------------
// E[e,m,o] = base[m,o] + 2.0 * sum_r h[e,m,r] * Bm[e,r,o]   (K=16 rank, VALU)
// ---------------------------------------------------------------------------
__global__ __launch_bounds__(256)
void lora_expand(const float* __restrict__ h, const float* __restrict__ Bm,
                 const float* __restrict__ base, float* __restrict__ E,
                 int M, int O)
{
    const int e = blockIdx.z;
    const size_t idx = (size_t)blockIdx.x * blockDim.x + threadIdx.x;
    if (idx >= (size_t)M * O) return;
    const int m = (int)(idx / O);
    const int o = (int)(idx - (size_t)m * O);
    const float* hp = h + ((size_t)e * M + m) * 16;
    const float* bp = Bm + (size_t)e * 16 * O + o;
    float s = 0.f;
    #pragma unroll
    for (int r = 0; r < 16; ++r) s += hp[r] * bp[(size_t)r * O];
    E[((size_t)e * M + m) * O + o] = base[idx] + 2.0f * s;
}

// ---------------------------------------------------------------------------
// Per-row (b,s) layernorm stats over the NE*O concatenated expert vector.
// ---------------------------------------------------------------------------
__global__ __launch_bounds__(256)
void ln_stats_kernel(const float* __restrict__ E, float* __restrict__ mu,
                     float* __restrict__ rstd, int O, int NO, int M)
{
    __shared__ float s1[256], s2[256];
    const int row = blockIdx.x;
    float a = 0.f, b = 0.f;
    for (int j = threadIdx.x; j < NO; j += 256) {
        const int e = j / O, o = j - e * O;
        const float v = E[((size_t)e * M + row) * O + o];
        a += v; b += v * v;
    }
    s1[threadIdx.x] = a; s2[threadIdx.x] = b;
    __syncthreads();
    for (int st = 128; st > 0; st >>= 1) {
        if (threadIdx.x < st) {
            s1[threadIdx.x] += s1[threadIdx.x + st];
            s2[threadIdx.x] += s2[threadIdx.x + st];
        }
        __syncthreads();
    }
    if (threadIdx.x == 0) {
        const float m = s1[0] / NO;
        const float var = s2[0] / NO - m * m;
        mu[row] = m;
        rstd[row] = rsqrtf(var + 1e-6f);
    }
}

__global__ void zero_kernel(float* p, int n)
{
    const int i = blockIdx.x * blockDim.x + threadIdx.x;
    if (i < n) p[i] = 0.f;
}

// ---------------------------------------------------------------------------
// logits[b,e'] += sum_{e,o} norm(Ecat[b,s,e*O+o]) * We[(s*NO + e*O+o), e']
// One block per (b,s); LDS reduce; atomic accumulate.
// ---------------------------------------------------------------------------
__global__ __launch_bounds__(256)
void gate_logits_kernel(const float* __restrict__ E, const float* __restrict__ mu,
                        const float* __restrict__ rstd, const float* __restrict__ g,
                        const float* __restrict__ bb, const float* __restrict__ We,
                        float* __restrict__ logits, int O, int NO, int S, int M)
{
    __shared__ float red[256 * 4];
    const int row = blockIdx.x;      // b*S + s
    const int b = row / S, s = row - b * S;
    const float m = mu[row], rs = rstd[row];
    float p0 = 0, p1 = 0, p2 = 0, p3 = 0;
    const float* Werow = We + (size_t)s * NO * 4;
    for (int j = threadIdx.x; j < NO; j += 256) {
        const int e = j / O, o = j - e * O;
        const float v = (E[((size_t)e * M + row) * O + o] - m) * rs * g[j] + bb[j];
        const float4 w = *(const float4*)(Werow + (size_t)j * 4);
        p0 += v * w.x; p1 += v * w.y; p2 += v * w.z; p3 += v * w.w;
    }
    red[threadIdx.x * 4 + 0] = p0; red[threadIdx.x * 4 + 1] = p1;
    red[threadIdx.x * 4 + 2] = p2; red[threadIdx.x * 4 + 3] = p3;
    __syncthreads();
    for (int st = 128; st > 0; st >>= 1) {
        if (threadIdx.x < st) {
            #pragma unroll
            for (int c = 0; c < 4; ++c)
                red[threadIdx.x * 4 + c] += red[(threadIdx.x + st) * 4 + c];
        }
        __syncthreads();
    }
    if (threadIdx.x < 4) atomicAdd(&logits[b * 4 + threadIdx.x], red[threadIdx.x]);
}

__global__ void gates_kernel(const float* __restrict__ logits,
                             const float* __restrict__ tau, float* __restrict__ gates)
{
    const int b = threadIdx.x;
    if (b >= 4) return;
    const float t = fmaxf(tau[0], 1e-6f);
    float l[4]; float mx = -1e30f;
    #pragma unroll
    for (int e = 0; e < 4; ++e) { l[e] = logits[b * 4 + e] / t; mx = fmaxf(mx, l[e]); }
    float sum = 0.f;
    #pragma unroll
    for (int e = 0; e < 4; ++e) { l[e] = expf(l[e] - mx); sum += l[e]; }
    #pragma unroll
    for (int e = 0; e < 4; ++e) gates[b * 4 + e] = l[e] / sum;
}

// ---------------------------------------------------------------------------
// x_out = base + sum_e gates[b,e]*E[e]   (+ optional RoPE), pair-per-thread.
// ---------------------------------------------------------------------------
__global__ __launch_bounds__(256)
void combine_rope_kernel(const float* __restrict__ base, const float* __restrict__ E,
                         const float* __restrict__ gates, const float* __restrict__ cs,
                         const float* __restrict__ sn, float* __restrict__ out,
                         int O, int M, int S, int doRope)
{
    const size_t idx = (size_t)blockIdx.x * blockDim.x + threadIdx.x;
    const size_t total = (size_t)M * (O / 2);
    if (idx >= total) return;
    const int row = (int)(idx / (O / 2));
    const int p = (int)(idx - (size_t)row * (O / 2));
    const int b = row / S, s = row - b * S;
    const int head = p >> 5, i = p & 31;            // HD/2 = 32
    const size_t bi = (size_t)row * O + head * 64 + 2 * i;
    const float g0 = gates[b * 4 + 0], g1 = gates[b * 4 + 1];
    const float g2 = gates[b * 4 + 2], g3 = gates[b * 4 + 3];
    const size_t MO = (size_t)M * O;
    float v0 = base[bi]     + g0 * E[bi]     + g1 * E[MO + bi]
             + g2 * E[2 * MO + bi]     + g3 * E[3 * MO + bi];
    float v1 = base[bi + 1] + g0 * E[bi + 1] + g1 * E[MO + bi + 1]
             + g2 * E[2 * MO + bi + 1] + g3 * E[3 * MO + bi + 1];
    if (doRope) {
        const float c = cs[s * 32 + i], ss = sn[s * 32 + i];
        out[bi]     = v0 * c - v1 * ss;
        out[bi + 1] = v0 * ss + v1 * c;
    } else {
        out[bi] = v0; out[bi + 1] = v1;
    }
}

// ---------------------------------------------------------------------------
// Causal GQA attention. One wave per (b, h, 16-row block).
// Scores tile 16 x S fp32 kept in LDS (64KB of the WGP's 320KB).
// Both Q@K^T and P@V run on v_wmma_f32_16x16x32_bf16.
// ---------------------------------------------------------------------------
__global__ __launch_bounds__(32)
void attn_kernel(const float* __restrict__ q, const float* __restrict__ k,
                 const float* __restrict__ v, float* __restrict__ out, int S)
{
    __shared__ float sc[16 * 1024];
    const int blk = blockIdx.x, h = blockIdx.y, b = blockIdx.z;
    const int lane = threadIdx.x;
    const int lm = lane & 15;
    const int kba = (lane >> 4) * 16;   // A-frag K base
    const int kbb = (lane >> 4) * 16;   // B-frag K base
    const int m0 = blk * 16;
    const int kvh = h >> 1;             // NREP = 2

    // Q fragments (A operand), scale folded in: 1/sqrt(64) = 0.125
    const float* qrow = q + ((size_t)b * S + m0 + lm) * 1024 + h * 64;
    v16bfT qf[2];
    #pragma unroll
    for (int kc = 0; kc < 2; ++kc)
        #pragma unroll
        for (int i = 0; i < 8; ++i) {
            qf[kc][i]     = (__bf16)(qrow[kc * 32 + kba + i] * 0.125f);
            qf[kc][i + 8] = (__bf16)(qrow[kc * 32 + kba + 8 + i] * 0.125f);
        }

    const int ntiles = blk + 1;          // causal: skip fully-masked key tiles
    for (int t = 0; t < ntiles; ++t) {
        const int t0 = t * 16;
        const float* krow = k + ((size_t)b * S + t0 + lm) * 512 + kvh * 64;
        v8fT accs = {};
        #pragma unroll
        for (int kc = 0; kc < 2; ++kc) {
            v16bfT kf;   // B operand: column = key lm, K = hd
            #pragma unroll
            for (int i = 0; i < 16; ++i) kf[i] = (__bf16)krow[kc * 32 + kbb + i];
            accs = __builtin_amdgcn_wmma_f32_16x16x32_bf16(false, qf[kc], false, kf,
                                                           (short)0, accs, false, false);
        }
        const int mb = (lane >> 4) * 8;
        #pragma unroll
        for (int r = 0; r < 8; ++r) {
            const int srow = m0 + mb + r, scol = t0 + lm;
            sc[(mb + r) * 1024 + scol] = accs[r] + (scol > srow ? -1e9f : 0.0f);
        }
    }
    __syncthreads();

    // Row-wise softmax over the valid prefix (lanes 0..15, one row each).
    const int L = ntiles * 16;
    if (lane < 16) {
        float* row = &sc[lane * 1024];
        float mx = -1e30f;
        for (int j = 0; j < L; ++j) mx = fmaxf(mx, row[j]);
        float sum = 0.f;
        for (int j = 0; j < L; ++j) { const float e = expf(row[j] - mx); row[j] = e; sum += e; }
        const float inv = 1.0f / sum;
        for (int j = 0; j < L; ++j) row[j] *= inv;
    }
    __syncthreads();

    // P @ V
    v8fT acc[4] = {{}, {}, {}, {}};
    for (int c0 = 0; c0 < L; c0 += 32) {
        v16bfT pf;   // A operand: row = query lm, K = key
        const float* prow = &sc[lm * 1024 + c0];
        #pragma unroll
        for (int i = 0; i < 8; ++i) {
            const int j0 = kba + i, j1 = kba + 8 + i;
            pf[i]     = (__bf16)((c0 + j0 < L) ? prow[j0] : 0.f);
            pf[i + 8] = (__bf16)((c0 + j1 < L) ? prow[j1] : 0.f);
        }
        #pragma unroll
        for (int nt = 0; nt < 4; ++nt) {
            v16bfT vf;   // B operand: column = hd (nt*16+lm), K = key
            #pragma unroll
            for (int i = 0; i < 16; ++i) {
                const int key = c0 + kbb + i;
                vf[i] = (__bf16)((key < L)
                    ? v[((size_t)b * S + key) * 512 + kvh * 64 + nt * 16 + lm] : 0.f);
            }
            acc[nt] = __builtin_amdgcn_wmma_f32_16x16x32_bf16(false, pf, false, vf,
                                                              (short)0, acc[nt], false, false);
        }
    }

    const int mb = (lane >> 4) * 8;
    #pragma unroll
    for (int nt = 0; nt < 4; ++nt)
        #pragma unroll
        for (int r = 0; r < 8; ++r)
            out[((size_t)b * S + m0 + mb + r) * 1024 + h * 64 + nt * 16 + lm] = acc[nt][r];
}

// ---------------------------------------------------------------------------
// Host launcher
// ---------------------------------------------------------------------------
extern "C" void kernel_launch(void* const* d_in, const int* in_sizes, int n_in,
                              void* d_out, int out_size, void* d_ws, size_t ws_size,
                              hipStream_t stream)
{
    (void)in_sizes; (void)n_in; (void)out_size; (void)ws_size;
    const int Bn = 4, S = 1024, D = 1024, M = Bn * S;   // 4096 rows
    const int NE = 4;

    const float* x    = (const float*)d_in[0];
    const float* wq   = (const float*)d_in[1];
    const float* wk   = (const float*)d_in[2];
    const float* wv   = (const float*)d_in[3];
    const float* wo   = (const float*)d_in[4];
    const float* cosb = (const float*)d_in[23];
    const float* sinb = (const float*)d_in[24];

    float* ws = (float*)d_ws;
    size_t off = 0;
    float* wbase = ws + off; off += (size_t)M * 1024;          // base (max O=1024)
    float* wh    = ws + off; off += (size_t)NE * M * 16;       // LoRA h
    float* wE    = ws + off; off += (size_t)NE * M * 1024;     // experts (max O)
    float* wxq   = ws + off; off += (size_t)M * 1024;
    float* wxk   = ws + off; off += (size_t)M * 512;
    float* wxv   = ws + off; off += (size_t)M * 512;
    float* wattn = ws + off; off += (size_t)M * 1024;
    float* wmu   = ws + off; off += M;
    float* wrstd = ws + off; off += M;
    float* wlog  = ws + off; off += 16;
    float* wgat  = ws + off; off += 16;

    struct Proj {
        const float* w; const float* A; const float* Bm; const float* g;
        const float* bb; const float* We; const float* tau;
        int O; float* xout; int doRope;
    };
    const Proj projs[3] = {
        { wq, (const float*)d_in[5],  (const float*)d_in[6],  (const float*)d_in[7],
          (const float*)d_in[8],  (const float*)d_in[9],  (const float*)d_in[10], 1024, wxq, 1 },
        { wk, (const float*)d_in[11], (const float*)d_in[12], (const float*)d_in[13],
          (const float*)d_in[14], (const float*)d_in[15], (const float*)d_in[16],  512, wxk, 1 },
        { wv, (const float*)d_in[17], (const float*)d_in[18], (const float*)d_in[19],
          (const float*)d_in[20], (const float*)d_in[21], (const float*)d_in[22],  512, wxv, 0 },
    };

    for (int pi = 0; pi < 3; ++pi) {
        const Proj& P = projs[pi];
        const int O = P.O, NO = NE * O;

        // base = x @ W
        gemm_bf16_wmma<<<dim3((O + 63) / 64, M / 64), 256, 0, stream>>>(
            x, P.w, wbase, M, O, D, D, O, O);

        // h_e = x @ A_e  (N = 16)
        for (int e = 0; e < NE; ++e)
            gemm_bf16_wmma<<<dim3(1, M / 64), 256, 0, stream>>>(
                x, P.A + (size_t)e * D * 16, wh + (size_t)e * M * 16,
                M, 16, D, D, 16, 16);

        // E_e = base + 2 * h_e @ B_e
        lora_expand<<<dim3(((size_t)M * O + 255) / 256, 1, NE), 256, 0, stream>>>(
            wh, P.Bm, wbase, wE, M, O);

        // layernorm stats over concatenated experts
        ln_stats_kernel<<<M, 256, 0, stream>>>(wE, wmu, wrstd, O, NO, M);

        // gating logits
        zero_kernel<<<1, 32, 0, stream>>>(wlog, 16);
        gate_logits_kernel<<<M, 256, 0, stream>>>(
            wE, wmu, wrstd, P.g, P.bb, P.We, wlog, O, NO, S, M);
        gates_kernel<<<1, 32, 0, stream>>>(wlog, P.tau, wgat);

        // x_out = base + sum_e g_e * E_e (+ RoPE)
        combine_rope_kernel<<<((size_t)M * (O / 2) + 255) / 256, 256, 0, stream>>>(
            wbase, wE, wgat, cosb, sinb, P.xout, O, M, S, P.doRope);
    }

    // attention (causal, GQA)
    attn_kernel<<<dim3(S / 16, 16, Bn), 32, 0, stream>>>(wxq, wxk, wxv, wattn, S);

    // out = attn @ wo
    gemm_bf16_wmma<<<dim3(1024 / 64, M / 64), 256, 0, stream>>>(
        wattn, wo, (float*)d_out, M, 1024, 1024, 1024, 1024, 1024);
}
